// CFChurn_DNN_89859305767609
// MI455X (gfx1250) — compile-verified, matrix operand order checked by reading, not run on hardware
//
#include <hip/hip_runtime.h>
#include <hip/hip_bf16.h>

typedef __attribute__((ext_vector_type(16))) _Float16 v16h;
typedef __attribute__((ext_vector_type(8)))  _Float16 v8h;
typedef __attribute__((ext_vector_type(8)))  float    v8f;

constexpr int NN = 50000;
constexpr int NE = 1600000;
constexpr int HH = 128;

// ---------------------------------------------------------------------------
// Pack a KxN f32 weight matrix into WMMA B-fragment order, fp16:
//   out[((kt*tiles_n + tn) << 9) + lane*16 + e]  holds W[k, n]
// where k = 16*(e>>3) + 8*(lane>>4) + (e&7) + 32*kt, n = tn*16 + (lane&15).
// Rows k >= Ksrc are zero-padded (used for the 16-row edge-attr weight).
// ---------------------------------------------------------------------------
__global__ void pack_w_kernel(const float* __restrict__ W, _Float16* __restrict__ out,
                              int Ksrc, int Kpad, int N) {
  long i = (long)blockIdx.x * blockDim.x + threadIdx.x;
  if (i >= (long)Kpad * N) return;
  const int k = (int)(i / N), n = (int)(i % N);
  const float v = (k < Ksrc) ? W[(long)k * N + n] : 0.0f;
  const int kt = k >> 5, kk = k & 31;
  const int eh = (kk >> 4) & 1, lh = (kk >> 3) & 1, el = kk & 7;
  const int e    = eh * 8 + el;
  const int lane = lh * 16 + (n & 15);
  const int tn   = n >> 4;
  out[(((long)kt * (N >> 4) + tn) << 9) + (lane << 4) + e] = (_Float16)v;
}

__device__ __forceinline__ v16h load_bfrag(const _Float16* __restrict__ p) {
  const v8h b0 = *(const v8h*)p;
  const v8h b1 = *(const v8h*)(p + 8);
  v16h b;
#pragma unroll
  for (int i = 0; i < 8; ++i) { b[i] = b0[i]; b[8 + i] = b1[i]; }
  return b;
}

// async 16B global -> LDS copy (GLOBAL_LOAD_ASYNC_TO_LDS_B128, ASYNCcnt)
__device__ __forceinline__ void async_g2l_b128(const void* gptr, void* lptr) {
  const unsigned lds_addr = (unsigned)(size_t)lptr;   // addr[31:0] == LDS offset
  asm volatile("global_load_async_to_lds_b128 %0, %1, off"
               :: "v"(lds_addr), "v"(gptr) : "memory");
}
__device__ __forceinline__ void wait_async0() {
  asm volatile("s_wait_asynccnt 0" ::: "memory");
}

// ---------------------------------------------------------------------------
// Generic tiled GEMM: Y[M,N] = act(X[M,K] @ W[K,N] + bias + res)
// 4 waves/block; each wave owns MT=4 consecutive 16-row tiles of one n-tile,
// so one B-fragment load feeds 4 WMMAs. f16 WMMA, f32 accumulate.
// ---------------------------------------------------------------------------
template<int ACT, int K>
__launch_bounds__(128)
__global__ void gemm_kernel(const float* __restrict__ X, const _Float16* __restrict__ Wp,
                            const float* __restrict__ bias, const float* __restrict__ res,
                            float* __restrict__ Y, int M, int N) {
  constexpr int MT = 4;
  const int lane = threadIdx.x & 31;
  const int wave = threadIdx.x >> 5;
  const int tiles_n = N >> 4;
  const int tiles_m = M >> 4;
  const int mgroups = (tiles_m + MT - 1) / MT;
  const long grp = (long)blockIdx.x * 4 + wave;
  if (grp >= (long)mgroups * tiles_n) return;
  const int tn  = (int)(grp % tiles_n);
  const int tm0 = (int)(grp / tiles_n) * MT;
  const int nn    = lane & 15;
  const int kbase = (lane >> 4) << 3;

  v8f acc[MT];
#pragma unroll
  for (int mt = 0; mt < MT; ++mt)
#pragma unroll
    for (int i = 0; i < 8; ++i) acc[mt][i] = 0.0f;

#pragma unroll
  for (int kt = 0; kt < K; kt += 32) {
    const v16h b = load_bfrag(Wp + (((long)(kt >> 5) * tiles_n + tn) << 9) + (lane << 4));
#pragma unroll
    for (int mt = 0; mt < MT; ++mt) {
      const int tm = tm0 + mt;
      if (tm < tiles_m) {               // uniform across the wave: EXEC stays all-ones
        const float* rowp = X + ((long)(tm << 4) + nn) * K + kt + kbase;
        const float4 f0 = *(const float4*)(rowp);
        const float4 f1 = *(const float4*)(rowp + 4);
        const float4 f2 = *(const float4*)(rowp + 16);
        const float4 f3 = *(const float4*)(rowp + 20);
        v16h a;
        a[0]  = (_Float16)f0.x; a[1]  = (_Float16)f0.y; a[2]  = (_Float16)f0.z; a[3]  = (_Float16)f0.w;
        a[4]  = (_Float16)f1.x; a[5]  = (_Float16)f1.y; a[6]  = (_Float16)f1.z; a[7]  = (_Float16)f1.w;
        a[8]  = (_Float16)f2.x; a[9]  = (_Float16)f2.y; a[10] = (_Float16)f2.z; a[11] = (_Float16)f2.w;
        a[12] = (_Float16)f3.x; a[13] = (_Float16)f3.y; a[14] = (_Float16)f3.z; a[15] = (_Float16)f3.w;
        acc[mt] = __builtin_amdgcn_wmma_f32_16x16x32_f16(false, a, false, b, (short)0, acc[mt],
                                                         false, false);
      }
    }
  }

  const int col = (tn << 4) + nn;
  const float bsum = bias ? bias[col] : 0.0f;
#pragma unroll
  for (int mt = 0; mt < MT; ++mt) {
    const int tm = tm0 + mt;
    if (tm < tiles_m) {
#pragma unroll
      for (int i = 0; i < 8; ++i) {
        const long m = ((long)tm << 4) + i + kbase;
        float v = acc[mt][i] + bsum;
        if (res) v += res[m * (long)N + col];
        if (ACT) v = fmaxf(v, 0.0f);
        Y[m * (long)N + col] = v;
      }
    }
  }
}

// ---------------------------------------------------------------------------
// e = relu( edge_attr @ W_edge + P[src] + Q[dst] + el_b ) -> fp16
// W_edge packed (zero-padded to K=32): single WMMA K-step.
// src/dst row indices fetched as int4 vectors; gathers use 32-bit offsets.
// ---------------------------------------------------------------------------
__launch_bounds__(128)
__global__ void edge_e_kernel(const float* __restrict__ eattr, const _Float16* __restrict__ Wp,
                              const float* __restrict__ bias, const float* __restrict__ P,
                              const float* __restrict__ Q, const int* __restrict__ src,
                              const int* __restrict__ dst, _Float16* __restrict__ e_out, int E) {
  const int lane = threadIdx.x & 31;
  const int wave = threadIdx.x >> 5;
  const long grp = (long)blockIdx.x * 4 + wave;
  if (grp >= (long)(E >> 4) * 8) return;
  const int  tn = (int)(grp & 7);
  const long tm = grp >> 3;
  const int nn    = lane & 15;
  const int kbase = (lane >> 4) << 3;

  const float* rowp = eattr + ((long)(tm << 4) + nn) * 16 + kbase;
  const float4 f0 = *(const float4*)(rowp);
  const float4 f1 = *(const float4*)(rowp + 4);
  v16h a;
  a[0] = (_Float16)f0.x; a[1] = (_Float16)f0.y; a[2] = (_Float16)f0.z; a[3] = (_Float16)f0.w;
  a[4] = (_Float16)f1.x; a[5] = (_Float16)f1.y; a[6] = (_Float16)f1.z; a[7] = (_Float16)f1.w;
#pragma unroll
  for (int i = 8; i < 16; ++i) a[i] = (_Float16)0.0f;  // K = 16..31 is zero padding

  const v16h b = load_bfrag(Wp + ((long)tn << 9) + (lane << 4));
  v8f acc;
#pragma unroll
  for (int i = 0; i < 8; ++i) acc[i] = 0.0f;
  acc = __builtin_amdgcn_wmma_f32_16x16x32_f16(false, a, false, b, (short)0, acc, false, false);

  const unsigned col = (unsigned)((tn << 4) + nn);
  const float bs = bias[col];
  const long mbase = (tm << 4) + kbase;
  const int4 s0 = *(const int4*)(src + mbase);
  const int4 s1 = *(const int4*)(src + mbase + 4);
  const int4 d0 = *(const int4*)(dst + mbase);
  const int4 d1 = *(const int4*)(dst + mbase + 4);
  const int ss[8] = {s0.x, s0.y, s0.z, s0.w, s1.x, s1.y, s1.z, s1.w};
  const int dd[8] = {d0.x, d0.y, d0.z, d0.w, d1.x, d1.y, d1.z, d1.w};
#pragma unroll
  for (int i = 0; i < 8; ++i) {
    const float v = acc[i] + P[(unsigned)ss[i] * 128u + col] + Q[(unsigned)dd[i] * 128u + col] + bs;
    e_out[(mbase + i) * 128 + col] = (_Float16)fmaxf(v, 0.0f);
  }
}

// ---------------------------------------------------------------------------
// Fused elconv message+aggregate: agg[dst] += e @ We + Hn[src]
// 256-thread block = 8 waves = the 8 n-tiles of one group of 4 edge tiles.
// Packed We (32KB) and the 4 e-tiles (16KB) are staged in LDS once per block
// via GLOBAL_LOAD_ASYNC_TO_LDS_B128 (ASYNCcnt), so the 410MB e tensor is read
// from HBM exactly once per layer and never bounces through VGPRs.
// ---------------------------------------------------------------------------
__launch_bounds__(256)
__global__ void edge_conv_kernel(const _Float16* __restrict__ e16, const _Float16* __restrict__ Wp,
                                 const float* __restrict__ Hn, const int* __restrict__ src,
                                 const int* __restrict__ dst, float* __restrict__ agg, int E) {
  constexpr int MT = 4;
  __shared__ __align__(32) _Float16 lds_w[4 * 8 * 512];   // 16384 halves = 32KB
  __shared__ __align__(32) _Float16 lds_a[64 * 128];      // 8192 halves  = 16KB
  const int lane = threadIdx.x & 31;
  const int tn   = threadIdx.x >> 5;                      // wave id == n-tile
  const int tm0  = blockIdx.x * MT;                       // grid = (E/16)/MT blocks
  const int tidb = threadIdx.x;

  // cooperative async stage: packed We then the 64 edge rows of e
#pragma unroll
  for (int i = 0; i < 8; ++i) {
    const int c = tidb + (i << 8);                        // 8 * 256 = 2048 chunks of 16B
    async_g2l_b128(Wp + ((long)c << 3), lds_w + (c << 3));
  }
  const _Float16* abase = e16 + ((long)tm0 << 4) * 128;
#pragma unroll
  for (int i = 0; i < 4; ++i) {
    const int c = tidb + (i << 8);                        // 4 * 256 = 1024 chunks of 16B
    async_g2l_b128(abase + ((long)c << 3), lds_a + (c << 3));
  }
  wait_async0();
  __syncthreads();

  const int nn    = lane & 15;
  const int kbase = (lane >> 4) << 3;

  v8f acc[MT];
#pragma unroll
  for (int mt = 0; mt < MT; ++mt)
#pragma unroll
    for (int i = 0; i < 8; ++i) acc[mt][i] = 0.0f;

#pragma unroll
  for (int kt = 0; kt < 128; kt += 32) {
    const _Float16* bp = lds_w + ((((kt >> 5) << 3) + tn) << 9) + (lane << 4);
    const v8h b0 = *(const v8h*)bp;
    const v8h b1 = *(const v8h*)(bp + 8);
    v16h b;
#pragma unroll
    for (int i = 0; i < 8; ++i) { b[i] = b0[i]; b[8 + i] = b1[i]; }
#pragma unroll
    for (int mt = 0; mt < MT; ++mt) {
      const _Float16* rp = lds_a + (((mt << 4) + nn) << 7) + kt + kbase;
      const v8h a0 = *(const v8h*)(rp);
      const v8h a1 = *(const v8h*)(rp + 16);
      v16h a;
#pragma unroll
      for (int i = 0; i < 8; ++i) { a[i] = a0[i]; a[8 + i] = a1[i]; }
      acc[mt] = __builtin_amdgcn_wmma_f32_16x16x32_f16(false, a, false, b, (short)0, acc[mt],
                                                       false, false);
    }
  }

  const unsigned col = (unsigned)((tn << 4) + nn);
#pragma unroll
  for (int mt = 0; mt < MT; ++mt) {
    const long mbase = ((long)(tm0 + mt) << 4) + kbase;
    const int4 s0 = *(const int4*)(src + mbase);
    const int4 s1 = *(const int4*)(src + mbase + 4);
    const int4 d0 = *(const int4*)(dst + mbase);
    const int4 d1 = *(const int4*)(dst + mbase + 4);
    const int ss[8] = {s0.x, s0.y, s0.z, s0.w, s1.x, s1.y, s1.z, s1.w};
    const int dd[8] = {d0.x, d0.y, d0.z, d0.w, d1.x, d1.y, d1.z, d1.w};
#pragma unroll
    for (int i = 0; i < 8; ++i) {
      const float v = acc[mt][i] + Hn[(unsigned)ss[i] * 128u + col];
      atomicAdd(&agg[(unsigned)dd[i] * 128u + col], v);
    }
  }
}

// ------------------------- scatter / elementwise ---------------------------
__global__ void zero_f32_kernel(float* p, long n) {
  long i = (long)blockIdx.x * blockDim.x + threadIdx.x;
  if (i < n) p[i] = 0.0f;
}
__global__ void degree_kernel(const int* __restrict__ dst, float* __restrict__ degcnt, int E) {
  int i = blockIdx.x * blockDim.x + threadIdx.x;
  if (i < E) atomicAdd(&degcnt[dst[i]], 1.0f);
}
__global__ void node_prep_kernel(const float* __restrict__ degcnt, float* __restrict__ dinv,
                                 float* __restrict__ deginv, float* __restrict__ cntinv, int N) {
  int i = blockIdx.x * blockDim.x + threadIdx.x;
  if (i >= N) return;
  const float c = degcnt[i];
  dinv[i]   = rsqrtf(c + 1.0f);
  deginv[i] = 1.0f / (c + 1.0f);
  cntinv[i] = 1.0f / fmaxf(c, 1.0f);
}
// agg[dst] += h[src] * dinv[src]*dinv[dst] ; one thread per (edge, 4 cols)
__global__ void gcn_agg_kernel(const float* __restrict__ h, const int* __restrict__ src,
                               const int* __restrict__ dst, const float* __restrict__ dinv,
                               float* __restrict__ agg, int E) {
  long tid = (long)blockIdx.x * blockDim.x + threadIdx.x;
  if (tid >= (long)E * 32) return;
  const unsigned c4 = (unsigned)(tid & 31) * 4u;
  const long e  = tid >> 5;
  const int s = src[e], d = dst[e];
  const float w = dinv[s] * dinv[d];
  const float4 hv = *(const float4*)(h + (unsigned)s * 128u + c4);
  float* ap = agg + (unsigned)d * 128u + c4;
  atomicAdd(ap + 0, hv.x * w);
  atomicAdd(ap + 1, hv.y * w);
  atomicAdd(ap + 2, hv.z * w);
  atomicAdd(ap + 3, hv.w * w);
}
__global__ void concat_dc_kernel(const float* __restrict__ xd, const float* __restrict__ xc,
                                 float* __restrict__ out, int N) {
  long i = (long)blockIdx.x * blockDim.x + threadIdx.x;
  if (i >= (long)N * 128) return;
  const unsigned n = (unsigned)(i >> 7), c = (unsigned)(i & 127);
  out[i] = (c < 32) ? xd[n * 38u + c] : xc[n * 96u + (c - 32)];
}
__global__ void gcn_fin_kernel(const float* __restrict__ agg, const float* __restrict__ h,
                               const float* __restrict__ deginv, const float* __restrict__ b,
                               float* __restrict__ out, int N) {
  long i = (long)blockIdx.x * blockDim.x + threadIdx.x;
  if (i >= (long)N * 128) return;
  const unsigned n = (unsigned)(i >> 7), c = (unsigned)(i & 127);
  out[i] = fmaxf(agg[i] + h[i] * deginv[n] + b[c], 0.0f);
}
__global__ void concat_x_kernel(const float* __restrict__ xdc, const float* __restrict__ g0,
                                const float* __restrict__ g1, float* __restrict__ out, int N) {
  long i = (long)blockIdx.x * blockDim.x + threadIdx.x;
  if (i >= (long)N * 256) return;
  const unsigned n = (unsigned)(i >> 8), c = (unsigned)(i & 255);
  out[i] = (c < 128) ? xdc[n * 128u + c]
                     : g0[n * 128u + (c - 128)] + g1[n * 128u + (c - 128)];
}
__global__ void elfin_kernel(const float* __restrict__ wr, const float* __restrict__ agg,
                             const float* __restrict__ cntinv, const float* __restrict__ b,
                             float* __restrict__ out, int N) {
  long i = (long)blockIdx.x * blockDim.x + threadIdx.x;
  if (i >= (long)N * 128) return;
  const unsigned n = (unsigned)(i >> 7), c = (unsigned)(i & 127);
  out[i] = fmaxf(wr[i] + agg[i] * cntinv[n] + b[c], 0.0f);
}
__global__ void add2_kernel(const float* __restrict__ a, const float* __restrict__ b,
                            float* __restrict__ out, long n) {
  long i = (long)blockIdx.x * blockDim.x + threadIdx.x;
  if (i < n) out[i] = a[i] + b[i];
}
__global__ void concat_h_kernel(const float* __restrict__ hci, const float* __restrict__ hsi,
                                float* __restrict__ out, int N) {
  long i = (long)blockIdx.x * blockDim.x + threadIdx.x;
  if (i >= (long)N * 256) return;
  const unsigned n = (unsigned)(i >> 8), c = (unsigned)(i & 255);
  out[i] = (c < 128) ? hci[n * 128u + c] : hsi[n * 128u + (c - 128)];
}
__global__ void softmax256_kernel(float* __restrict__ a, int N) {
  int i = blockIdx.x * blockDim.x + threadIdx.x;
  if (i >= N) return;
  float* row = a + (long)i * 256;
  float mx = -1e30f;
  for (int k = 0; k < 256; ++k) mx = fmaxf(mx, row[k]);
  float s = 0.0f;
  for (int k = 0; k < 256; ++k) { const float v = __expf(row[k] - mx); row[k] = v; s += v; }
  const float inv = 1.0f / s;
  for (int k = 0; k < 256; ++k) row[k] *= inv;
}
__global__ void attn_comb_kernel(const float* __restrict__ a, const float* __restrict__ hci,
                                 const float* __restrict__ hsi, float* __restrict__ py, int N) {
  long i = (long)blockIdx.x * blockDim.x + threadIdx.x;
  if (i >= (long)N * 128) return;
  const unsigned n = (unsigned)(i >> 7), c = (unsigned)(i & 127);
  py[i] = a[n * 256u + c] * hci[i] + a[n * 256u + 128u + c] * hsi[i];
}
__global__ void vecdot_sig_kernel(const float* __restrict__ X, const float* __restrict__ w,
                                  const float* __restrict__ b, float* __restrict__ out, int N) {
  int i = blockIdx.x * blockDim.x + threadIdx.x;
  if (i >= N) return;
  float s = b[0];
  const float4* xr = (const float4*)(X + (long)i * 128);
  const float4* wr = (const float4*)w;
#pragma unroll 4
  for (int k = 0; k < 32; ++k) {
    const float4 xv = xr[k], wv = wr[k];
    s += xv.x * wv.x + xv.y * wv.y + xv.z * wv.z + xv.w * wv.w;
  }
  out[i] = 1.0f / (1.0f + __expf(-s));
}
__global__ void final_kernel(const float* __restrict__ y0, const float* __restrict__ y1,
                             const int* __restrict__ t, float* __restrict__ oy,
                             float* __restrict__ ocf, int N) {
  int i = blockIdx.x * blockDim.x + threadIdx.x;
  if (i >= N) return;
  const float tf = (float)t[i];
  oy[i]  = (1.0f - tf) * y0[i] + tf * y1[i];
  ocf[i] = tf * y0[i] + (1.0f - tf) * y1[i];
}

// ---------------------------------------------------------------------------
extern "C" void kernel_launch(void* const* d_in, const int* in_sizes, int n_in,
                              void* d_out, int out_size, void* d_ws, size_t ws_size,
                              hipStream_t stream) {
  (void)in_sizes; (void)n_in; (void)out_size; (void)ws_size;
  const float* discrete_x  = (const float*)d_in[0];
  const float* continous_x = (const float*)d_in[1];
  const float* edge_attr   = (const float*)d_in[2];
  const float* W_c    = (const float*)d_in[3];  const float* b_c    = (const float*)d_in[4];
  const float* W_g0   = (const float*)d_in[5];  const float* b_g0   = (const float*)d_in[6];
  const float* gcn1_W = (const float*)d_in[7];  const float* gcn1_b = (const float*)d_in[8];
  const float* gcn2_W = (const float*)d_in[9];  const float* gcn2_b = (const float*)d_in[10];
  const float* el_W   = (const float*)d_in[11]; const float* el_b   = (const float*)d_in[12];
  const float* res_W1 = (const float*)d_in[13]; const float* res_b1 = (const float*)d_in[14];
  const float* res_W2 = (const float*)d_in[15]; const float* res_b2 = (const float*)d_in[16];
  const float* fus_W  = (const float*)d_in[17]; const float* fus_b  = (const float*)d_in[18];
  const float* si0_W  = (const float*)d_in[19]; const float* si0_b  = (const float*)d_in[20];
  const float* c1_Wn  = (const float*)d_in[21]; const float* c1_We  = (const float*)d_in[22];
  const float* c1_Wr  = (const float*)d_in[23]; const float* c1_b   = (const float*)d_in[24];
  const float* c2_Wn  = (const float*)d_in[25]; const float* c2_We  = (const float*)d_in[26];
  const float* c2_Wr  = (const float*)d_in[27]; const float* c2_b   = (const float*)d_in[28];
  const float* at0_W  = (const float*)d_in[29]; const float* at0_b  = (const float*)d_in[30];
  const float* at1_W  = (const float*)d_in[31]; const float* at1_b  = (const float*)d_in[32];
  const float* y0_W1  = (const float*)d_in[33]; const float* y0_b1  = (const float*)d_in[34];
  const float* y0_W2  = (const float*)d_in[35]; const float* y0_b2  = (const float*)d_in[36];
  const float* y1_W1  = (const float*)d_in[37]; const float* y1_b1  = (const float*)d_in[38];
  const float* y1_W2  = (const float*)d_in[39]; const float* y1_b2  = (const float*)d_in[40];
  const float* T_W    = (const float*)d_in[41]; const float* T_b    = (const float*)d_in[42];
  const int* edge_index = (const int*)d_in[43];
  const int* t          = (const int*)d_in[44];
  const int* src = edge_index;
  const int* dst = edge_index + NE;

  float* out  = (float*)d_out;
  float* o_y   = out;
  float* o_cf  = out + (long)NN;
  float* o_y0  = out + 2L * NN;
  float* o_y1  = out + 3L * NN;
  float* o_T   = out + 4L * NN;
  float* o_hci = out + 5L * NN;
  float* o_hsi = out + 5L * NN + (long)NN * HH;

  // ---- workspace carve-up (bump allocator, 64B aligned) ----
  char* wsb = (char*)d_ws;
  size_t off = 0;
  auto af = [&](size_t n) -> float* {
    off = (off + 63) & ~(size_t)63; float* p = (float*)(wsb + off); off += n * 4; return p; };
  auto ah = [&](size_t n) -> _Float16* {
    off = (off + 63) & ~(size_t)63; _Float16* p = (_Float16*)(wsb + off); off += n * 2; return p; };

  _Float16* w_c   = ah(32 * 32);
  _Float16* w_g0  = ah(128 * 128);
  _Float16* w_gc1 = ah(128 * 128);
  _Float16* w_gc2 = ah(128 * 128);
  _Float16* w_elS = ah(128 * 128);
  _Float16* w_elD = ah(128 * 128);
  _Float16* w_elE = ah(32 * 128);     // zero-padded to K=32
  _Float16* w_r1  = ah(256 * 256);
  _Float16* w_r2  = ah(256 * 256);
  _Float16* w_fus = ah(256 * 128);
  _Float16* w_si  = ah(256 * 128);
  _Float16* w_c1n = ah(128 * 128);
  _Float16* w_c1e = ah(128 * 128);
  _Float16* w_c1r = ah(128 * 128);
  _Float16* w_c2n = ah(128 * 128);
  _Float16* w_c2e = ah(128 * 128);
  _Float16* w_c2r = ah(128 * 128);
  _Float16* w_a0  = ah(256 * 256);
  _Float16* w_a1  = ah(256 * 256);
  _Float16* w_y0  = ah(128 * 128);
  _Float16* w_y1  = ah(128 * 128);
  _Float16* e16   = ah((size_t)NE * 128);   // fp16 edge features (410MB)

  float* x_c    = af((size_t)NN * 96);
  float* x_dc   = af((size_t)NN * 128);
  float* x_g    = af((size_t)NN * 128);
  float* hbuf   = af((size_t)NN * 128);
  float* agg    = af((size_t)NN * 128);
  float* x_g0   = af((size_t)NN * 128);
  float* x_g1   = af((size_t)NN * 128);
  float* degcnt = af(NN);
  float* dinv   = af(NN);
  float* deginv = af(NN);
  float* cntinv = af(NN);
  float* xcat   = af((size_t)NN * 256);
  float* h1     = af((size_t)NN * 256);
  float* x_deep = af((size_t)NN * 256);
  float* x_si   = af((size_t)NN * 128);
  float* Pb     = af((size_t)NN * 128);
  float* Qb     = af((size_t)NN * 128);
  float* Hn     = af((size_t)NN * 128);
  float* wr     = af((size_t)NN * 128);
  float* x_si0  = af((size_t)NN * 128);
  float* x_si1  = af((size_t)NN * 128);
  float* hcat   = af((size_t)NN * 256);
  float* a0     = af((size_t)NN * 256);
  float* a1     = af((size_t)NN * 256);
  float* py0    = af((size_t)NN * 128);
  float* py1    = af((size_t)NN * 128);
  float* z0     = af((size_t)NN * 128);
  float* z1     = af((size_t)NN * 128);

  auto pack = [&](const float* s, _Float16* dp, int Ksrc, int Kpad, int N) {
    const long tot = (long)Kpad * N;
    pack_w_kernel<<<(int)((tot + 255) / 256), 256, 0, stream>>>(s, dp, Ksrc, Kpad, N); };
  auto zeroN128 = [&](float* p) {
    zero_f32_kernel<<<(int)(((long)NN * 128 + 255) / 256), 256, 0, stream>>>(p, (long)NN * 128); };
  auto gg = [](long M, long N) -> unsigned {   // MT = 4 m-tiles per wave, 4 waves/block
    const long g = (((M >> 4) + 3) / 4) * (N >> 4); return (unsigned)((g + 3) / 4); };
  const int EW = 128;
  const unsigned grid_ee = (unsigned)(((long)(NE / 16) * 8 + 3) / 4);
  const unsigned grid_ec = (unsigned)(NE / 64);            // 256-thread LDS blocks

  // ---- weight packing (f32 -> fp16 WMMA fragment order) ----
  pack(W_c, w_c, 32, 32, 32);
  pack(W_g0, w_g0, 128, 128, 128);
  pack(gcn1_W, w_gc1, 128, 128, 128);
  pack(gcn2_W, w_gc2, 128, 128, 128);
  pack(el_W, w_elS, 128, 128, 128);
  pack(el_W + 128 * 128, w_elD, 128, 128, 128);
  pack(el_W + 256 * 128, w_elE, 16, 32, 128);   // rows 16..31 zeroed
  pack(res_W1, w_r1, 256, 256, 256);
  pack(res_W2, w_r2, 256, 256, 256);
  pack(fus_W, w_fus, 256, 256, 128);
  pack(si0_W, w_si, 256, 256, 128);
  pack(c1_Wn, w_c1n, 128, 128, 128); pack(c1_We, w_c1e, 128, 128, 128); pack(c1_Wr, w_c1r, 128, 128, 128);
  pack(c2_Wn, w_c2n, 128, 128, 128); pack(c2_We, w_c2e, 128, 128, 128); pack(c2_Wr, w_c2r, 128, 128, 128);
  pack(at0_W, w_a0, 256, 256, 256);
  pack(at1_W, w_a1, 256, 256, 256);
  pack(y0_W1, w_y0, 128, 128, 128);
  pack(y1_W1, w_y1, 128, 128, 128);

  // ---- degrees ----
  zero_f32_kernel<<<(NN + 255) / 256, 256, 0, stream>>>(degcnt, NN);
  degree_kernel<<<(NE + 255) / 256, 256, 0, stream>>>(dst, degcnt, NE);
  node_prep_kernel<<<(NN + 255) / 256, 256, 0, stream>>>(degcnt, dinv, deginv, cntinv, NN);

  // ---- featurization ----
  gemm_kernel<1, 32><<<gg(3L * NN, 32), EW, 0, stream>>>(continous_x, w_c, b_c, nullptr,
                                                         x_c, 3 * NN, 32);
  concat_dc_kernel<<<(int)(((long)NN * 128 + 255) / 256), 256, 0, stream>>>(discrete_x, x_c, x_dc, NN);
  gemm_kernel<1, 128><<<gg(NN, 128), EW, 0, stream>>>(x_dc, w_g0, b_g0, nullptr, x_g, NN, 128);

  // ---- GCN layer 1 ----
  gemm_kernel<0, 128><<<gg(NN, 128), EW, 0, stream>>>(x_g, w_gc1, nullptr, nullptr, hbuf, NN, 128);
  zeroN128(agg);
  gcn_agg_kernel<<<(int)(((long)NE * 32 + 255) / 256), 256, 0, stream>>>(hbuf, src, dst, dinv, agg, NE);
  gcn_fin_kernel<<<(int)(((long)NN * 128 + 255) / 256), 256, 0, stream>>>(agg, hbuf, deginv, gcn1_b, x_g0, NN);
  // ---- GCN layer 2 ----
  gemm_kernel<0, 128><<<gg(NN, 128), EW, 0, stream>>>(x_g0, w_gc2, nullptr, nullptr, hbuf, NN, 128);
  zeroN128(agg);
  gcn_agg_kernel<<<(int)(((long)NE * 32 + 255) / 256), 256, 0, stream>>>(hbuf, src, dst, dinv, agg, NE);
  gcn_fin_kernel<<<(int)(((long)NN * 128 + 255) / 256), 256, 0, stream>>>(agg, hbuf, deginv, gcn2_b, x_g1, NN);

  // ---- trunk ----
  concat_x_kernel<<<(int)(((long)NN * 256 + 255) / 256), 256, 0, stream>>>(x_dc, x_g0, x_g1, xcat, NN);
  gemm_kernel<1, 256><<<gg(NN, 256), EW, 0, stream>>>(xcat, w_r1, res_b1, nullptr, h1, NN, 256);
  gemm_kernel<0, 256><<<gg(NN, 256), EW, 0, stream>>>(h1, w_r2, res_b2, xcat, x_deep, NN, 256);
  gemm_kernel<1, 256><<<gg(NN, 128), EW, 0, stream>>>(x_deep, w_fus, fus_b, nullptr, o_hci, NN, 128);
  gemm_kernel<1, 256><<<gg(NN, 128), EW, 0, stream>>>(xcat, w_si, si0_b, nullptr, x_si, NN, 128);

  // ---- edge features: el_W split into src/dst/edge parts ----
  gemm_kernel<0, 128><<<gg(NN, 128), EW, 0, stream>>>(x_si, w_elS, nullptr, nullptr, Pb, NN, 128);
  gemm_kernel<0, 128><<<gg(NN, 128), EW, 0, stream>>>(x_si, w_elD, nullptr, nullptr, Qb, NN, 128);
  edge_e_kernel<<<grid_ee, EW, 0, stream>>>(edge_attr, w_elE, el_b, Pb, Qb, src, dst, e16, NE);

  // ---- elconv layer 1 ----
  gemm_kernel<0, 128><<<gg(NN, 128), EW, 0, stream>>>(x_si, w_c1n, nullptr, nullptr, Hn, NN, 128);
  zeroN128(agg);
  edge_conv_kernel<<<grid_ec, 256, 0, stream>>>(e16, w_c1e, Hn, src, dst, agg, NE);
  gemm_kernel<0, 128><<<gg(NN, 128), EW, 0, stream>>>(x_si, w_c1r, nullptr, nullptr, wr, NN, 128);
  elfin_kernel<<<(int)(((long)NN * 128 + 255) / 256), 256, 0, stream>>>(wr, agg, cntinv, c1_b, x_si0, NN);
  // ---- elconv layer 2 ----
  gemm_kernel<0, 128><<<gg(NN, 128), EW, 0, stream>>>(x_si0, w_c2n, nullptr, nullptr, Hn, NN, 128);
  zeroN128(agg);
  edge_conv_kernel<<<grid_ec, 256, 0, stream>>>(e16, w_c2e, Hn, src, dst, agg, NE);
  gemm_kernel<0, 128><<<gg(NN, 128), EW, 0, stream>>>(x_si0, w_c2r, nullptr, nullptr, wr, NN, 128);
  elfin_kernel<<<(int)(((long)NN * 128 + 255) / 256), 256, 0, stream>>>(wr, agg, cntinv, c2_b, x_si1, NN);

  add2_kernel<<<(int)(((long)NN * 128 + 255) / 256), 256, 0, stream>>>(x_si0, x_si1, o_hsi, (long)NN * 128);
  vecdot_sig_kernel<<<(NN + 255) / 256, 256, 0, stream>>>(o_hsi, T_W, T_b, o_T, NN);

  // ---- attention heads ----
  concat_h_kernel<<<(int)(((long)NN * 256 + 255) / 256), 256, 0, stream>>>(o_hci, o_hsi, hcat, NN);
  gemm_kernel<0, 256><<<gg(NN, 256), EW, 0, stream>>>(hcat, w_a0, at0_b, nullptr, a0, NN, 256);
  gemm_kernel<0, 256><<<gg(NN, 256), EW, 0, stream>>>(hcat, w_a1, at1_b, nullptr, a1, NN, 256);
  softmax256_kernel<<<(NN + 255) / 256, 256, 0, stream>>>(a0, NN);
  softmax256_kernel<<<(NN + 255) / 256, 256, 0, stream>>>(a1, NN);
  attn_comb_kernel<<<(int)(((long)NN * 128 + 255) / 256), 256, 0, stream>>>(a0, o_hci, o_hsi, py0, NN);
  attn_comb_kernel<<<(int)(((long)NN * 128 + 255) / 256), 256, 0, stream>>>(a1, o_hci, o_hsi, py1, NN);

  gemm_kernel<1, 128><<<gg(NN, 128), EW, 0, stream>>>(py0, w_y0, y0_b1, nullptr, z0, NN, 128);
  gemm_kernel<1, 128><<<gg(NN, 128), EW, 0, stream>>>(py1, w_y1, y1_b1, nullptr, z1, NN, 128);
  vecdot_sig_kernel<<<(NN + 255) / 256, 256, 0, stream>>>(z0, y0_W2, y0_b2, o_y0, NN);
  vecdot_sig_kernel<<<(NN + 255) / 256, 256, 0, stream>>>(z1, y1_W2, y1_b2, o_y1, NN);
  final_kernel<<<(NN + 255) / 256, 256, 0, stream>>>(o_y0, o_y1, t, o_y, o_cf, NN);
}
